// SLiCE_15891378995308
// MI455X (gfx1250) — matrix operand order repaired and checked by reading, not compiled, not run on hardware
//
#include <hip/hip_runtime.h>
#include <hip/hip_bf16.h>

// ---------------------------------------------------------------------------
// SLiCE forward for MI455X (gfx1250):
//   bf16 WMMA (hi/lo split, 3-product compensation) + TDM async X staging
//   + chunked parallel scan (3 passes).
// ---------------------------------------------------------------------------

typedef __attribute__((ext_vector_type(16))) __bf16 v16bf;
typedef __attribute__((ext_vector_type(8)))  float  v8f;
typedef __attribute__((ext_vector_type(4)))  unsigned int tdm_u32x4;
typedef __attribute__((ext_vector_type(8)))  int          tdm_i32x8;
typedef __attribute__((ext_vector_type(4)))  int          tdm_i32x4;

#define B_DIM      8
#define T_DIM      8192
#define D_IN       128
#define HID        128
#define NBLK       32            // 128 / 4
#define KPAD       160           // 129 padded to 5 * 32
#define NOUT       640           // 512 (W_A) + 128 (W_B)
#define KT_TILES   5             // KPAD / 32
#define NT_TILES   40            // NOUT / 16
#define TC         32            // timesteps per chunk
#define NC         (T_DIM / TC)  // 256 chunks per batch
#define MT_TILES   (TC / 16)     // 2
#define THREADS    256
#define NWAVES     (THREADS / 32)
#define FRAG_ELEMS (KT_TILES * NT_TILES * 32 * 16)  // ushorts per weight plane

// LDS layout (dynamic): Ab f32 [TC][NOUT] | inph/inpl bf16 [TC][KPAD] | xstage f32 [TC+1][D_IN]
#define LDS_AB_BYTES   (TC * NOUT * 4)              // 81920
#define LDS_INP_BYTES  (TC * KPAD * 2)              // 10240 each plane
#define LDS_XS_OFF     (LDS_AB_BYTES + 2 * LDS_INP_BYTES)
#define LDS_XS_BYTES   ((TC + 1) * D_IN * 4)        // 16896
#define SMEM_BYTES     (LDS_XS_OFF + LDS_XS_BYTES)  // 119296

#if defined(__has_builtin)
#if __has_builtin(__builtin_amdgcn_tensor_load_to_lds)
#define USE_TDM 1
#endif
#endif

static __device__ __forceinline__ unsigned short f2bf(float f) {
  union { float f; unsigned u; } v; v.f = f;
  unsigned r = v.u + 0x7fffu + ((v.u >> 16) & 1u);  // round-to-nearest-even
  return (unsigned short)(r >> 16);
}
static __device__ __forceinline__ float bf2f(unsigned short h) {
  union { float f; unsigned u; } v; v.u = ((unsigned)h) << 16;
  return v.f;
}

// ---------------------------------------------------------------------------
// Pass 0: pre-swizzle [W_A | W_B] into WMMA B-operand fragments (hi/lo bf16).
// Fragment layout: [(kt*NT + nt)*32 + lane] * 16 contiguous bf16 per lane.
// B (32x16 KxN) lane mapping: n = lane&15, K-base = (lane>>4)*8, elems j<8: K=base+j, j>=8: K=base+8+j.
// ---------------------------------------------------------------------------
__global__ void slice_prep_w(const float* __restrict__ WA,
                             const float* __restrict__ WB,
                             unsigned short* __restrict__ wb_hi,
                             unsigned short* __restrict__ wb_lo) {
  int tid = blockIdx.x * blockDim.x + threadIdx.x;
  if (tid >= KT_TILES * NT_TILES * 32) return;
  int kt   = tid / (NT_TILES * 32);
  int rem  = tid % (NT_TILES * 32);
  int nt   = rem / 32;
  int lane = rem % 32;
  int n     = nt * 16 + (lane & 15);
  int khalf = kt * 32 + (lane >> 4) * 8;
  size_t base = (size_t)tid * 16;
  for (int j = 0; j < 16; ++j) {
    int k = khalf + (j < 8 ? j : 8 + j);
    float w = 0.0f;
    if (k < D_IN + 1) w = (n < 512) ? WA[k * 512 + n] : WB[k * 128 + (n - 512)];
    unsigned short hi = f2bf(w);
    unsigned short lo = f2bf(w - bf2f(hi));
    wb_hi[base + j] = hi;
    wb_lo[base + j] = lo;
  }
}

// ---------------------------------------------------------------------------
// Stage X rows [t0-1 .. t0+TC-1] of batch b into LDS xstage[0..TC][128]
// via the Tensor Data Mover (2D D#: tile = nrows x 128 f32). Row for g=-1
// (chunk 0 only) is zero-filled by the workgroup. TENSORcnt-tracked.
// ---------------------------------------------------------------------------
static __device__ void stage_x_chunk(const float* __restrict__ X,
                                     float* __restrict__ xstage,
                                     int b, int c) {
  const int tid    = threadIdx.x;
  const int t0     = c * TC;
  const int gstart = (c == 0) ? 0 : t0 - 1;
  const int nrows  = (c == 0) ? TC : TC + 1;
  const int row0   = (c == 0) ? 1 : 0;   // LDS row where staged data begins

  if (c == 0) {
    for (int i = tid; i < D_IN; i += THREADS) xstage[i] = 0.0f;  // g = -1 row
  }

  const float* gsrc = X + ((size_t)b * T_DIM + gstart) * D_IN;

#ifdef USE_TDM
  if (tid < 32) {  // one wave issues the DMA (EXEC is ignored by TDM)
    unsigned long long gaddr = (unsigned long long)(uintptr_t)gsrc;
    unsigned lds_off = (unsigned)(uintptr_t)(void*)(xstage + row0 * D_IN);

    tdm_u32x4 g0;
    g0[0] = 1u;                                   // count=1, user descriptor
    g0[1] = lds_off;                              // lds_addr (bytes)
    g0[2] = (unsigned)(gaddr & 0xffffffffull);    // global_addr[31:0]
    g0[3] = (unsigned)((gaddr >> 32) & 0x01ffffffull) | (2u << 30);  // addr[56:32] | type=2

    tdm_i32x8 g1;
    g1[0] = (int)(2u << 16);                      // wg_mask=0 | data_size=2 (4 bytes)
    g1[1] = (int)((unsigned)D_IN << 16);          // tensor_dim0[15:0] = 128
    g1[2] = (int)(((unsigned)nrows & 0xffffu) << 16); // dim0[31:16]=0 | tensor_dim1[15:0]
    g1[3] = (int)((unsigned)D_IN << 16);          // tensor_dim1[31:16]=0 | tile_dim0 = 128
    g1[4] = nrows;                                // tile_dim1 | tile_dim2=0
    g1[5] = D_IN;                                 // tensor_dim0_stride[31:0] = 128
    g1[6] = 0;
    g1[7] = 0;

    tdm_i32x4 gz4 = {0, 0, 0, 0};
#if __clang_major__ >= 23
    tdm_i32x8 gz8 = {0, 0, 0, 0, 0, 0, 0, 0};
    __builtin_amdgcn_tensor_load_to_lds(g0, g1, gz4, gz4, gz8, 0);
#else
    __builtin_amdgcn_tensor_load_to_lds(g0, g1, gz4, gz4, 0);
#endif
    __builtin_amdgcn_s_wait_tensorcnt(0);
  }
#else
  for (int idx = tid; idx < nrows * D_IN; idx += THREADS)
    xstage[row0 * D_IN + idx] = gsrc[idx];
#endif
  __syncthreads();
}

// ---------------------------------------------------------------------------
// Shared: compute Ab = inp @ [W_A|W_B] for one (batch, chunk) into LDS (f32).
// inp row t: [1, X[g]-X[g-1]] (from xstage), zero-padded to KPAD, hi/lo bf16.
// 3-product bf16 split per K-tile: ah*bh + ah*bl + al*bh  (~fp32 accuracy).
// Each wave owns an nt column and feeds BOTH mt accumulators per B-fragment.
// ---------------------------------------------------------------------------
static __device__ void chunk_gemm(const unsigned short* __restrict__ wb_hi,
                                  const unsigned short* __restrict__ wb_lo,
                                  float* __restrict__ Ab,            // LDS
                                  unsigned short* __restrict__ inph, // LDS
                                  unsigned short* __restrict__ inpl, // LDS
                                  const float* __restrict__ xstage)  // LDS
{
  const int tid = threadIdx.x;

  // Warm L2 for the weight-fragment region (global_prefetch_b8).
  {
    const char* wp = (const char*)wb_hi;
    size_t bytes = (size_t)FRAG_ELEMS * 2u * sizeof(unsigned short);
    for (size_t off = (size_t)tid * 1024; off < bytes; off += (size_t)THREADS * 1024)
      __builtin_prefetch(wp + off, 0, 2);
  }

  // Build inp (hi/lo bf16) in LDS from staged X rows.
  for (int idx = tid; idx < TC * KPAD; idx += THREADS) {
    int t = idx / KPAD, k = idx % KPAD;
    float v = 0.0f;
    if (k == 0) {
      v = 1.0f;
    } else if (k <= D_IN) {
      v = xstage[(t + 1) * D_IN + (k - 1)] - xstage[t * D_IN + (k - 1)];
    }
    unsigned short hi = f2bf(v);
    unsigned short lo = f2bf(v - bf2f(hi));
    inph[idx] = hi;
    inpl[idx] = lo;
  }
  __syncthreads();

  const int lane  = tid & 31;
  const int wave  = tid >> 5;
  const int row16 = lane & 15;        // A row within tile / D column
  const int khi8  = (lane >> 4) * 8;  // A-operand K sub-offset per lane half

  for (int nt = wave; nt < NT_TILES; nt += NWAVES) {
    v8f acc0 = {};
    v8f acc1 = {};
    for (int kt = 0; kt < KT_TILES; ++kt) {
      size_t boff = (((size_t)(kt * NT_TILES + nt)) * 32 + lane) * 16;
      v16bf bh = *(const v16bf*)(wb_hi + boff);
      v16bf bl = *(const v16bf*)(wb_lo + boff);
      int kb = kt * 32 + khi8;
      // A layout (16-bit): lanes 0-15 K=0..7,16..23; lanes 16-31 K=8..15,24..31.
#pragma unroll
      for (int mt = 0; mt < MT_TILES; ++mt) {
        union { v16bf v; unsigned short u[16]; } ah, al;
        int rbase = (mt * 16 + row16) * KPAD;
#pragma unroll
        for (int j = 0; j < 8; ++j) {
          ah.u[j]     = inph[rbase + kb + j];
          al.u[j]     = inpl[rbase + kb + j];
          ah.u[8 + j] = inph[rbase + kb + 16 + j];
          al.u[8 + j] = inpl[rbase + kb + 16 + j];
        }
        if (mt == 0) {
          acc0 = __builtin_amdgcn_wmma_f32_16x16x32_bf16(false, ah.v, false, bh, (short)0, acc0, false, false);
          acc0 = __builtin_amdgcn_wmma_f32_16x16x32_bf16(false, ah.v, false, bl, (short)0, acc0, false, false);
          acc0 = __builtin_amdgcn_wmma_f32_16x16x32_bf16(false, al.v, false, bh, (short)0, acc0, false, false);
        } else {
          acc1 = __builtin_amdgcn_wmma_f32_16x16x32_bf16(false, ah.v, false, bh, (short)0, acc1, false, false);
          acc1 = __builtin_amdgcn_wmma_f32_16x16x32_bf16(false, ah.v, false, bl, (short)0, acc1, false, false);
          acc1 = __builtin_amdgcn_wmma_f32_16x16x32_bf16(false, al.v, false, bh, (short)0, acc1, false, false);
        }
      }
    }
    // D layout: VGPR v -> row M = v + 8*(lane>>4), col N = lane&15.
    int colb = nt * 16 + (lane & 15);
    int rb0  = 8 * (lane >> 4);
#pragma unroll
    for (int vv = 0; vv < 8; ++vv) {
      Ab[(rb0 + vv) * NOUT + colb]      = acc0[vv];
      Ab[(16 + rb0 + vv) * NOUT + colb] = acc1[vv];
    }
  }
  __syncthreads();
}

// ---------------------------------------------------------------------------
// Pass 1: per-(batch,chunk) GEMM + chunk-total affine map per 4x4 block.
// Column-parallel composition: thread (blk, col<4) carries column col of M_cum,
// thread (blk, 4) carries b_cum. No cross-thread traffic (m read from LDS).
// ---------------------------------------------------------------------------
__global__ void slice_pass1(const float* __restrict__ X,
                            const unsigned short* __restrict__ wb_hi,
                            const unsigned short* __restrict__ wb_lo,
                            float* __restrict__ totals) {
  extern __shared__ char smem[];
  float*          Ab     = (float*)smem;
  unsigned short* inph   = (unsigned short*)(smem + LDS_AB_BYTES);
  unsigned short* inpl   = inph + TC * KPAD;
  float*          xstage = (float*)(smem + LDS_XS_OFF);
  int c = blockIdx.x, b = blockIdx.y;

  stage_x_chunk(X, xstage, b, c);
  chunk_gemm(wb_hi, wb_lo, Ab, inph, inpl, xstage);

  int tid = threadIdx.x;
  if (tid < NBLK * 5) {
    int blk = tid / 5;
    int col = tid % 5;  // 0..3 = matrix columns, 4 = bias column
    float vec[4];
#pragma unroll
    for (int i = 0; i < 4; ++i) vec[i] = (col < 4 && i == col) ? 1.0f : 0.0f;
    for (int t = 0; t < TC; ++t) {
      const float* row = Ab + t * NOUT;
      float m[4][4];
#pragma unroll
      for (int i = 0; i < 4; ++i)
#pragma unroll
        for (int l = 0; l < 4; ++l)
          m[i][l] = row[blk * 16 + i * 4 + l] + ((i == l) ? 1.0f : 0.0f);
      float nv[4];
#pragma unroll
      for (int i = 0; i < 4; ++i) {
        float s = 0.0f;
#pragma unroll
        for (int l = 0; l < 4; ++l) s += m[i][l] * vec[l];
        if (col == 4) s += row[512 + blk * 4 + i];
        nv[i] = s;
      }
#pragma unroll
      for (int i = 0; i < 4; ++i) vec[i] = nv[i];
    }
    float* dst = totals + (((size_t)(b * NC + c)) * NBLK + blk) * 20;
    if (col < 4) {
#pragma unroll
      for (int i = 0; i < 4; ++i) dst[i * 4 + col] = vec[i];
    } else {
#pragma unroll
      for (int i = 0; i < 4; ++i) dst[16 + i] = vec[i];
    }
  }
}

// ---------------------------------------------------------------------------
// Pass 2: tiny sequential scan of chunk totals -> chunk-entry states y_in.
// One thread per (batch, block) = 256 threads, one workgroup.
// ---------------------------------------------------------------------------
__global__ void slice_pass2(const float* __restrict__ totals,
                            const float* __restrict__ y0,
                            float* __restrict__ ycarry) {
  int tid = threadIdx.x;
  int b = tid >> 5, blk = tid & 31;
  float y[4];
#pragma unroll
  for (int i = 0; i < 4; ++i) y[i] = y0[blk * 4 + i];
  for (int c = 0; c < NC; ++c) {
    size_t e = ((size_t)(b * NC + c)) * NBLK + blk;
    float* dst = ycarry + e * 4;
#pragma unroll
    for (int i = 0; i < 4; ++i) dst[i] = y[i];
    const float* src = totals + e * 20;
    float ny[4];
#pragma unroll
    for (int i = 0; i < 4; ++i) {
      float s = src[16 + i];
#pragma unroll
      for (int j = 0; j < 4; ++j) s += src[i * 4 + j] * y[j];
      ny[i] = s;
    }
#pragma unroll
    for (int i = 0; i < 4; ++i) y[i] = ny[i];
  }
}

// ---------------------------------------------------------------------------
// Pass 3: recompute chunk GEMM, run the 32-step recurrence from y_in, emit y.
// ---------------------------------------------------------------------------
__global__ void slice_pass3(const float* __restrict__ X,
                            const unsigned short* __restrict__ wb_hi,
                            const unsigned short* __restrict__ wb_lo,
                            const float* __restrict__ ycarry,
                            float* __restrict__ out) {
  extern __shared__ char smem[];
  float*          Ab     = (float*)smem;
  unsigned short* inph   = (unsigned short*)(smem + LDS_AB_BYTES);
  unsigned short* inpl   = inph + TC * KPAD;
  float*          xstage = (float*)(smem + LDS_XS_OFF);
  int c = blockIdx.x, b = blockIdx.y;

  stage_x_chunk(X, xstage, b, c);
  chunk_gemm(wb_hi, wb_lo, Ab, inph, inpl, xstage);

  int tid = threadIdx.x;
  if (tid < NBLK) {
    int blk = tid;
    const float* ysrc = ycarry + (((size_t)(b * NC + c)) * NBLK + blk) * 4;
    float y[4];
#pragma unroll
    for (int i = 0; i < 4; ++i) y[i] = ysrc[i];
    int t0 = c * TC;
    for (int t = 0; t < TC; ++t) {
      const float* row = Ab + t * NOUT;
      float ny[4];
#pragma unroll
      for (int i = 0; i < 4; ++i) {
        float s = row[512 + blk * 4 + i];
#pragma unroll
        for (int l = 0; l < 4; ++l)
          s += (row[blk * 16 + i * 4 + l] + ((i == l) ? 1.0f : 0.0f)) * y[l];
        ny[i] = s;
      }
#pragma unroll
      for (int i = 0; i < 4; ++i) y[i] = ny[i];
      size_t g = (size_t)b * T_DIM + (t0 + t);
      float4 o = make_float4(y[0], y[1], y[2], y[3]);
      *(float4*)(out + g * HID + blk * 4) = o;
    }
  }
}

// ---------------------------------------------------------------------------
extern "C" void kernel_launch(void* const* d_in, const int* in_sizes, int n_in,
                              void* d_out, int out_size, void* d_ws, size_t ws_size,
                              hipStream_t stream) {
  const float* X  = (const float*)d_in[0];
  const float* WA = (const float*)d_in[1];
  const float* WB = (const float*)d_in[2];
  const float* y0 = (const float*)d_in[3];
  float* out = (float*)d_out;

  char* ws = (char*)d_ws;
  unsigned short* wb_hi = (unsigned short*)ws;
  unsigned short* wb_lo = wb_hi + FRAG_ELEMS;
  float* totals = (float*)(ws + (size_t)2 * FRAG_ELEMS * sizeof(unsigned short));
  float* ycarry = totals + (size_t)B_DIM * NC * NBLK * 20;
  // ws usage: 0.41 MB fragments + 5.24 MB totals + 1.05 MB ycarry ~= 6.7 MB

  slice_prep_w<<<(KT_TILES * NT_TILES * 32 + 255) / 256, 256, 0, stream>>>(WA, WB, wb_hi, wb_lo);

  dim3 grid(NC, B_DIM);  // 2048 workgroups
  slice_pass1<<<grid, THREADS, SMEM_BYTES, stream>>>(X, wb_hi, wb_lo, totals);
  slice_pass2<<<1, 256, 0, stream>>>(totals, y0, ycarry);
  slice_pass3<<<grid, THREADS, SMEM_BYTES, stream>>>(X, wb_hi, wb_lo, ycarry, out);

  (void)in_sizes; (void)n_in; (void)out_size; (void)ws_size;
}
